// AttentionalFactorizationLayer_6975026889271
// MI455X (gfx1250) — matrix operand order repaired and checked by reading, not compiled, not run on hardware
//
#include <hip/hip_runtime.h>
#include <math.h>

typedef __attribute__((ext_vector_type(16))) _Float16 v16h;
typedef __attribute__((ext_vector_type(8)))  float    v8f;

static constexpr int kF      = 50;                  // NUM_FIELDS
static constexpr int kD      = 16;                  // EMBED_DIM == ATTN_SIZE
static constexpr int kP      = (kF * (kF - 1)) / 2; // 1225 pairs
static constexpr int kTiles  = 80;                  // padded: 4 waves * 20 tiles
static constexpr int kPPad   = kTiles * 16;         // 1280
static constexpr int kBlock  = 128;
static constexpr int kWaves  = kBlock / 32;
static constexpr int kRowB   = kD * (int)sizeof(float);   // 64-byte row stride

__device__ __forceinline__ int pairs_before(int i) {
    return i * (kF - 1) - (i * (i - 1)) / 2;
}

// ---- single-instruction max (skip IEEE canonicalize; data is NaN-free) -----
__device__ __forceinline__ float relu_fast(float v) {
    float r;
    asm("v_max_num_f32 %0, %1, 0" : "=v"(r) : "v"(v));
    return r;
}
__device__ __forceinline__ float max_fast(float a, float b) {
    float r;
    asm("v_max_num_f32 %0, %1, %2" : "=v"(r) : "v"(a), "v"(b));
    return r;
}

// ---- cross-lane helpers (pure VALU) ----------------------------------------
template <int CTRL>
__device__ __forceinline__ float dpp_bcast(float v) {
    int p = __builtin_amdgcn_update_dpp(0, __float_as_int(v), CTRL, 0xF, 0xF, true);
    return __int_as_float(p);
}
__device__ __forceinline__ float row_sum16(float v) {
    v += dpp_bcast<0xB1>(v);   // xor 1
    v += dpp_bcast<0x4E>(v);   // xor 2
    v += dpp_bcast<0x141>(v);  // xor 7 (row_half_mirror)
    v += dpp_bcast<0x140>(v);  // xor 15 (row_mirror)
    return v;
}
__device__ __forceinline__ float row_max16(float v) {
    v = max_fast(v, dpp_bcast<0xB1>(v));
    v = max_fast(v, dpp_bcast<0x4E>(v));
    v = max_fast(v, dpp_bcast<0x141>(v));
    v = max_fast(v, dpp_bcast<0x140>(v));
    return v;
}
__device__ __forceinline__ float permx16(float v) {   // lane <-> lane^16
    int r = __builtin_amdgcn_permlanex16(__float_as_int(v), __float_as_int(v),
                                         0x76543210u, 0xFEDCBA98u, true, false);
    return __int_as_float(r);
}

// ---- one 16-pair tile, transposed GEMM: D' = W^T @ ip^T (+bias in C) -------
// d[k] = h[p = lane&15, a = k + 8*grp]; projection is per-lane + one xor16.
__device__ __forceinline__ void score_tile(
    int tile, const float* x_s, const unsigned* pairs_s, float* scores_s,
    const v16h& amatW, const v8f& cbias, const float* pwv, float pb, int m, int grp)
{
    v16h bvec;                                   // ip^T, B layout
    if (grp == 0) {                              // lanes 0-15 carry K=0..15
        const unsigned pk = pairs_s[tile * 16 + m];   // pre-scaled byte offsets
        const float* xi = (const float*)((const char*)x_s + (pk >> 16));
        const float* xj = (const float*)((const char*)x_s + (pk & 0xffffu));
        #pragma unroll
        for (int e = 0; e < 16; ++e)
            bvec[e] = (_Float16)(xi[e] * xj[e]);
    } else {                                     // lanes 16-31: K=16..31 = 0
        #pragma unroll
        for (int e = 0; e < 16; ++e) bvec[e] = (_Float16)0.0f;
    }

    v8f d = __builtin_amdgcn_wmma_f32_16x16x32_f16(
                false, amatW, false, bvec, (short)0, cbias, false, false);

    float t = 0.0f;
    #pragma unroll
    for (int k = 0; k < 8; ++k)
        t = fmaf(relu_fast(d[k]), pwv[k], t);
    t += permx16(t);                             // combine a=0..7 and a=8..15 halves
    const int pp = tile * 16 + m;                // lane l and l^16: same value, same addr
    scores_s[pp] = (pp < kP) ? (t + pb) : -INFINITY;
}

__global__ __launch_bounds__(kBlock) void afm_attention_kernel(
    const float* __restrict__ x,        // [B, F, D]
    const float* __restrict__ attn_W,   // [16,16]
    const float* __restrict__ attn_b,   // [16]
    const float* __restrict__ proj_W,   // [16]
    const float* __restrict__ proj_b,   // [1]
    const float* __restrict__ fc_W,     // [16]
    const float* __restrict__ fc_b,     // [1]
    float* __restrict__ out)            // [B]
{
    __shared__ float    x_s[kF * kD];
    __shared__ unsigned pairs_s[kPPad];
    __shared__ float    scores_s[kPPad];
    __shared__ float    pooled_s[kD];
    __shared__ float    red_s[2 * kWaves];
    __shared__ float    bcast_s[2];

    const int tid  = threadIdx.x;
    const int lane = tid & 31;
    const int wv   = tid >> 5;
    const int m    = lane & 15;
    const int grp  = lane >> 4;
    const int b    = blockIdx.x;
    const float* xb = x + (size_t)b * kF * kD;

    // ---- init: stage x[b], zero pooled, build padded pair table ----
    for (int t = tid; t < kF * kD; t += kBlock) x_s[t] = xb[t];
    if (tid < kD) pooled_s[tid] = 0.0f;
    for (int p = tid; p < kPPad; p += kBlock) {
        unsigned pk = 0u;                      // padded -> (0,0): benign
        if (p < kP) {
            int i = (int)floorf((99.0f - sqrtf(9801.0f - 8.0f * (float)p)) * 0.5f);
            if (i < 0) i = 0;
            while (pairs_before(i + 1) <= p) ++i;
            while (pairs_before(i) > p) --i;
            int j = i + 1 + (p - pairs_before(i));
            pk = ((unsigned)(i * kRowB) << 16) | (unsigned)(j * kRowB);  // byte offsets
        }
        pairs_s[p] = pk;
    }

    // ---- A operand: W^T in A layout; per-lane bias / proj-weight vectors ---
    // A'[M=a, K=d] = attn_W[d*16 + a]; lane m holds a=m, K = 16*grp + e (e<16)
    v16h amatW;
    #pragma unroll
    for (int e = 0; e < 16; ++e) {
        amatW[e] = (grp == 0) ? (_Float16)attn_W[e * 16 + m] : (_Float16)0.0f;  // K = 16..31 zero padding
    }
    v8f cbias;                                   // C[a,p] = attn_b[a]
    float pwv[8];
    #pragma unroll
    for (int k = 0; k < 8; ++k) {
        cbias[k] = attn_b[8 * grp + k];
        pwv[k]   = proj_W[8 * grp + k];
    }
    const float pb = proj_b[0];

    __syncthreads();

    // ---- pass 1: scores, two independent tiles per iteration ----
    for (int t0 = wv * 2; t0 < kTiles; t0 += kWaves * 2) {
        score_tile(t0,     x_s, pairs_s, scores_s, amatW, cbias, pwv, pb, m, grp);
        score_tile(t0 + 1, x_s, pairs_s, scores_s, amatW, cbias, pwv, pb, m, grp);
    }
    __syncthreads();

    // ---- softmax: max (DPP within 16-groups, LDS across 8 groups) ----
    float lmax = -INFINITY;
    for (int idx = tid; idx < kPPad; idx += kBlock) lmax = max_fast(lmax, scores_s[idx]);
    lmax = row_max16(lmax);
    if (m == 0) red_s[wv * 2 + grp] = lmax;
    __syncthreads();
    if (tid == 0) {
        float mxv = red_s[0];
        #pragma unroll
        for (int w = 1; w < 2 * kWaves; ++w) mxv = max_fast(mxv, red_s[w]);
        bcast_s[0] = mxv;
    }
    __syncthreads();
    const float mx = bcast_s[0];

    // ---- softmax: exp + sum (exp stored in place) ----
    float lsum = 0.0f;
    for (int idx = tid; idx < kPPad; idx += kBlock) {
        float e = __expf(scores_s[idx] - mx);   // padded: exp(-inf) = 0
        scores_s[idx] = e;
        lsum += e;
    }
    lsum = row_sum16(lsum);
    if (m == 0) red_s[wv * 2 + grp] = lsum;
    __syncthreads();
    if (tid == 0) {
        float s = 0.0f;
        #pragma unroll
        for (int w = 0; w < 2 * kWaves; ++w) s += red_s[w];
        bcast_s[1] = 1.0f / s;
    }
    __syncthreads();
    const float invZ = bcast_s[1];

    // ---- pass 2: attention-weighted pooling (f32 recompute of ip) ----
    float acc[8];
    #pragma unroll
    for (int k = 0; k < 8; ++k) acc[k] = 0.0f;
    for (int tile = wv; tile < kTiles; tile += kWaves) {
        const int p = tile * 16 + m;
        const unsigned pk = pairs_s[p];
        const float* xi = (const float*)((const char*)x_s + (pk >> 16));
        const float* xj = (const float*)((const char*)x_s + (pk & 0xffffu));
        const int dbase = grp * 8;
        const float w = scores_s[p] * invZ;     // 0 for padded pairs
        #pragma unroll
        for (int k = 0; k < 8; ++k) {
            float v = xi[dbase + k] * xj[dbase + k];
            acc[k] = fmaf(w, v, acc[k]);
        }
    }
    #pragma unroll
    for (int k = 0; k < 8; ++k) {
        float t = row_sum16(acc[k]);            // DPP, no LDS
        if (m == 0) atomicAdd(&pooled_s[grp * 8 + k], t);  // once per wave
    }
    __syncthreads();

    // ---- final FC: pooled @ fc_W + fc_b ----
    if (tid == 0) {
        float r = fc_b[0];
        #pragma unroll
        for (int d = 0; d < kD; ++d) r = fmaf(pooled_s[d], fc_W[d], r);
        out[b] = r;
    }
}

extern "C" void kernel_launch(void* const* d_in, const int* in_sizes, int n_in,
                              void* d_out, int out_size, void* d_ws, size_t ws_size,
                              hipStream_t stream) {
    const float* x      = (const float*)d_in[0];
    const float* attn_W = (const float*)d_in[1];
    const float* attn_b = (const float*)d_in[2];
    const float* proj_W = (const float*)d_in[3];
    const float* proj_b = (const float*)d_in[4];
    const float* fc_W   = (const float*)d_in[5];
    const float* fc_b   = (const float*)d_in[6];
    float* out = (float*)d_out;

    const int B = in_sizes[0] / (kF * kD);   // 8192
    afm_attention_kernel<<<B, kBlock, 0, stream>>>(
        x, attn_W, attn_b, proj_W, proj_b, fc_W, fc_b, out);
}
